// MHA_876173329020
// MI455X (gfx1250) — matrix operand order repaired and checked by reading, not compiled, not run on hardware
//
#include <hip/hip_runtime.h>
#include <hip/hip_bf16.h>

// Problem constants (match reference)
#define BB 4
#define LL 2048
#define EE 1024
#define HH 16
#define DD 64

typedef __attribute__((ext_vector_type(16))) __bf16 v16bf;
typedef __attribute__((ext_vector_type(8)))  float  v8f;

#define LDS_STRIDE 40  // 32 data bf16 + 8 pad -> 80B row, keeps 16B alignment

__device__ __forceinline__ v8f wmma_bf16(v16bf a, v16bf b, v8f c) {
    return __builtin_amdgcn_wmma_f32_16x16x32_bf16(
        /*neg_a=*/false, a, /*neg_b=*/false, b,
        /*c_mod=*/(short)0, c, /*reuse_a=*/false, /*reuse_b=*/false);
}

__device__ __forceinline__ void cvt_hilo(float x, __bf16& h, __bf16& l) {
    h = (__bf16)x;
    l = (__bf16)(x - (float)h);
}

// Prefetch a [128 x 32] fp32 tile (row stride srcStride) into 4 float4 regs/thread.
__device__ __forceinline__ void tile_load_g(const float* __restrict__ src, int srcStride,
                                            int tid, float4 (&r)[4]) {
#pragma unroll
    for (int t = 0; t < 4; t++) {
        int idx = tid + (t << 8);
        int rr = idx >> 3;
        int c  = (idx & 7) << 2;
        r[t] = *(const float4*)(src + (size_t)rr * srcStride + c);
    }
}

// Convert prefetched regs to hi/lo bf16 and store into LDS tile.
__device__ __forceinline__ void tile_cvt_store(const float4 (&r)[4],
                                               __bf16* __restrict__ dh,
                                               __bf16* __restrict__ dl, int tid) {
#pragma unroll
    for (int t = 0; t < 4; t++) {
        int idx = tid + (t << 8);
        int rr = idx >> 3;
        int c  = (idx & 7) << 2;
        int o = rr * LDS_STRIDE + c;
        __bf16 h, l;
        cvt_hilo(r[t].x, h, l); dh[o + 0] = h; dl[o + 0] = l;
        cvt_hilo(r[t].y, h, l); dh[o + 1] = h; dl[o + 1] = l;
        cvt_hilo(r[t].z, h, l); dh[o + 2] = h; dl[o + 2] = l;
        cvt_hilo(r[t].w, h, l); dh[o + 3] = h; dl[o + 3] = l;
    }
}

// A-fragment (16x32, 16-bit): lane m holds K{0..7,16..23}; lane m+16 holds K{8..15,24..31}
__device__ __forceinline__ v16bf frag_ld_a(const __bf16* s, int row, int lane) {
    int l = lane & 15;
    int k = (lane & 16) ? 8 : 0;
    const __bf16* p = s + (row + l) * LDS_STRIDE + k;
    v16bf r;
#pragma unroll
    for (int i = 0; i < 8; i++) r[i] = p[i];
#pragma unroll
    for (int i = 0; i < 8; i++) r[i + 8] = p[16 + i];
    return r;
}

// B-fragment (32x16, 16-bit): lane n = column n, K{0..7,16..23}; lane n+16 = column n, K{8..15,24..31}
__device__ __forceinline__ v16bf frag_ld_b(const __bf16* s, int row, int lane) {
    int l = lane & 15;
    int k = (lane & 16) ? 8 : 0;
    const __bf16* p = s + (row + l) * LDS_STRIDE + k;
    v16bf r;
#pragma unroll
    for (int i = 0; i < 8; i++) r[i] = p[i];
#pragma unroll
    for (int i = 0; i < 8; i++) r[i + 8] = p[16 + i];
    return r;
}

// ---------------------------------------------------------------------------
// Generic GEMM: C[M x 1024] = A[M x K] @ W[1024 x K]^T  (torch Linear)
// headSplit=1 -> write C in [b, h, l, d] layout; else plain row-major [M,1024].
// Block tile 128x128, 8 waves (4x2 of 32x64 wave tiles), K-step 32,
// software-pipelined global->reg prefetch of the next K-slice.
// ---------------------------------------------------------------------------
__global__ __launch_bounds__(256) void gemm_xwT_kernel(
    const float* __restrict__ A, const float* __restrict__ W,
    float* __restrict__ C, int M, int K, int headSplit) {
    __shared__ __bf16 sAh[128 * LDS_STRIDE];
    __shared__ __bf16 sAl[128 * LDS_STRIDE];
    __shared__ __bf16 sBh[128 * LDS_STRIDE];
    __shared__ __bf16 sBl[128 * LDS_STRIDE];

    const int tid  = threadIdx.x;
    const int lane = tid & 31;
    const int wave = tid >> 5;
    const int m0 = blockIdx.x * 128;
    const int n0 = blockIdx.y * 128;
    const int wm = (wave >> 1) * 32;  // 0,32,64,96
    const int wn = (wave & 1) * 64;   // 0,64

    const float* Abase = A + (size_t)m0 * K;
    const float* Wbase = W + (size_t)n0 * K;

    v8f acc[2][4];
#pragma unroll
    for (int i = 0; i < 2; i++)
#pragma unroll
        for (int j = 0; j < 4; j++)
#pragma unroll
            for (int r = 0; r < 8; r++) acc[i][j][r] = 0.0f;

    float4 pa[4], pb[4];
    tile_load_g(Abase, K, tid, pa);
    tile_load_g(Wbase, K, tid, pb);
    tile_cvt_store(pa, sAh, sAl, tid);
    tile_cvt_store(pb, sBh, sBl, tid);
    __syncthreads();

    for (int kt = 0; kt < K; kt += 32) {
        const bool hasNext = (kt + 32) < K;
        if (hasNext) {  // issue next tile's global loads; latency hidden by WMMAs
            tile_load_g(Abase + kt + 32, K, tid, pa);
            tile_load_g(Wbase + kt + 32, K, tid, pb);
        }

        v16bf ah[2], al[2], bh[4], bl[4];
#pragma unroll
        for (int mi = 0; mi < 2; mi++) {
            ah[mi] = frag_ld_a(sAh, wm + mi * 16, lane);
            al[mi] = frag_ld_a(sAl, wm + mi * 16, lane);
        }
#pragma unroll
        for (int nj = 0; nj < 4; nj++) {
            bh[nj] = frag_ld_b(sBh, wn + nj * 16, lane);
            bl[nj] = frag_ld_b(sBl, wn + nj * 16, lane);
        }
#pragma unroll
        for (int mi = 0; mi < 2; mi++)
#pragma unroll
            for (int nj = 0; nj < 4; nj++) {
                acc[mi][nj] = wmma_bf16(ah[mi], bh[nj], acc[mi][nj]);
                acc[mi][nj] = wmma_bf16(ah[mi], bl[nj], acc[mi][nj]);
                acc[mi][nj] = wmma_bf16(al[mi], bh[nj], acc[mi][nj]);
            }
        __syncthreads();
        if (hasNext) {
            tile_cvt_store(pa, sAh, sAl, tid);
            tile_cvt_store(pb, sBh, sBl, tid);
        }
        __syncthreads();
    }

    const int lcol  = lane & 15;
    const int rbase = (lane & 16) ? 8 : 0;
#pragma unroll
    for (int mi = 0; mi < 2; mi++)
#pragma unroll
        for (int nj = 0; nj < 4; nj++)
#pragma unroll
            for (int r = 0; r < 8; r++) {
                int gm = m0 + wm + mi * 16 + rbase + r;
                int gn = n0 + wn + nj * 16 + lcol;
                float val = acc[mi][nj][r];
                size_t o;
                if (headSplit) {
                    int b = gm >> 11;          // / LL
                    int lpos = gm & (LL - 1);
                    int h = gn >> 6, d = gn & 63;
                    o = (((size_t)(b * HH + h)) * LL + lpos) * DD + d;
                } else {
                    o = (size_t)gm * EE + gn;
                }
                C[o] = val;
            }
}

// ---------------------------------------------------------------------------
// Scores: S[b,h, i, j] = (Q . K) / sqrt(D) for j-tiles <= i-tile (causal).
// Raw scores written; softmax kernel handles the mask region.
// ---------------------------------------------------------------------------
__global__ __launch_bounds__(256) void attn_scores_kernel(
    const float* __restrict__ Qb, const float* __restrict__ Kb,
    float* __restrict__ attn) {
    const int it = blockIdx.x, jt = blockIdx.y, bh = blockIdx.z;
    if (jt > it) return;

    __shared__ __bf16 sAh[128 * LDS_STRIDE];
    __shared__ __bf16 sAl[128 * LDS_STRIDE];
    __shared__ __bf16 sBh[128 * LDS_STRIDE];
    __shared__ __bf16 sBl[128 * LDS_STRIDE];

    const float* Q  = Qb + (size_t)bh * LL * DD + (size_t)it * 128 * DD;
    const float* Km = Kb + (size_t)bh * LL * DD + (size_t)jt * 128 * DD;
    float* S = attn + (size_t)bh * LL * LL;

    const int tid = threadIdx.x, lane = tid & 31, wave = tid >> 5;
    const int wm = (wave >> 1) * 32;
    const int wn = (wave & 1) * 64;

    v8f acc[2][4];
#pragma unroll
    for (int i = 0; i < 2; i++)
#pragma unroll
        for (int j = 0; j < 4; j++)
#pragma unroll
            for (int r = 0; r < 8; r++) acc[i][j][r] = 0.0f;

    float4 pa[4], pb[4];
    tile_load_g(Q, DD, tid, pa);
    tile_load_g(Km, DD, tid, pb);
    tile_cvt_store(pa, sAh, sAl, tid);
    tile_cvt_store(pb, sBh, sBl, tid);
    __syncthreads();

    for (int kt = 0; kt < DD; kt += 32) {
        const bool hasNext = (kt + 32) < DD;
        if (hasNext) {
            tile_load_g(Q + kt + 32, DD, tid, pa);
            tile_load_g(Km + kt + 32, DD, tid, pb);
        }
        v16bf ah[2], al[2], bh[4], bl[4];
#pragma unroll
        for (int mi = 0; mi < 2; mi++) {
            ah[mi] = frag_ld_a(sAh, wm + mi * 16, lane);
            al[mi] = frag_ld_a(sAl, wm + mi * 16, lane);
        }
#pragma unroll
        for (int nj = 0; nj < 4; nj++) {
            bh[nj] = frag_ld_b(sBh, wn + nj * 16, lane);
            bl[nj] = frag_ld_b(sBl, wn + nj * 16, lane);
        }
#pragma unroll
        for (int mi = 0; mi < 2; mi++)
#pragma unroll
            for (int nj = 0; nj < 4; nj++) {
                acc[mi][nj] = wmma_bf16(ah[mi], bh[nj], acc[mi][nj]);
                acc[mi][nj] = wmma_bf16(ah[mi], bl[nj], acc[mi][nj]);
                acc[mi][nj] = wmma_bf16(al[mi], bh[nj], acc[mi][nj]);
            }
        __syncthreads();
        if (hasNext) {
            tile_cvt_store(pa, sAh, sAl, tid);
            tile_cvt_store(pb, sBh, sBl, tid);
        }
        __syncthreads();
    }

    const float scale = 0.125f;  // 1/sqrt(64)
    const int lcol = lane & 15;
    const int rbase = (lane & 16) ? 8 : 0;
#pragma unroll
    for (int mi = 0; mi < 2; mi++)
#pragma unroll
        for (int nj = 0; nj < 4; nj++)
#pragma unroll
            for (int r = 0; r < 8; r++) {
                int gi = it * 128 + wm + mi * 16 + rbase + r;
                int gj = jt * 128 + wn + nj * 16 + lcol;
                S[(size_t)gi * LL + gj] = acc[mi][nj][r] * scale;
            }
}

// ---------------------------------------------------------------------------
// Row softmax with causal bound. One block per row; each thread caches its
// 8 row elements in registers -> single global read pass + single write pass
// (writes probs for j <= i, zeros for j > i, completing the attn output).
// ---------------------------------------------------------------------------
__global__ __launch_bounds__(256) void softmax_kernel(float* __restrict__ attn) {
    __shared__ float red[256];
    const size_t row = blockIdx.x;
    const int i = (int)(row & (LL - 1));
    float* p = attn + row * (size_t)LL;
    const int n = i + 1;
    const int tid = threadIdx.x;

    float vals[8];  // LL / 256 == 8
    float m = -3.0e38f;
#pragma unroll
    for (int t = 0; t < 8; t++) {
        int j = tid + (t << 8);
        vals[t] = (j < n) ? p[j] : -3.0e38f;
        m = fmaxf(m, vals[t]);
    }
    red[tid] = m;
    __syncthreads();
    for (int s = 128; s > 0; s >>= 1) {
        if (tid < s) red[tid] = fmaxf(red[tid], red[tid + s]);
        __syncthreads();
    }
    m = red[0];
    __syncthreads();

    float sum = 0.0f;
#pragma unroll
    for (int t = 0; t < 8; t++) {
        int j = tid + (t << 8);
        float e = (j < n) ? __expf(vals[t] - m) : 0.0f;
        vals[t] = e;
        sum += e;
    }
    red[tid] = sum;
    __syncthreads();
    for (int s = 128; s > 0; s >>= 1) {
        if (tid < s) red[tid] += red[tid + s];
        __syncthreads();
    }
    const float inv = 1.0f / red[0];

#pragma unroll
    for (int t = 0; t < 8; t++) {
        int j = tid + (t << 8);
        p[j] = (j < n) ? vals[t] * inv : 0.0f;
    }
}

// ---------------------------------------------------------------------------
// OH[b, l, (h d)] = P[b,h] @ V[b,h]  (only j-tiles <= i-tile; rest is zero)
// Block tile 128(i) x 64(d); 8 waves of 16 rows each; K-step 32 over j,
// with register prefetch of the next P/V slices.
// ---------------------------------------------------------------------------
__global__ __launch_bounds__(256) void attn_av_kernel(
    const float* __restrict__ attn, const float* __restrict__ Vb,
    float* __restrict__ OH) {
    const int it = blockIdx.x, bh = blockIdx.y;
    const int b = bh / HH, h = bh % HH;
    const float* P = attn + (size_t)bh * LL * LL + (size_t)it * 128 * LL;
    const float* V = Vb + (size_t)bh * LL * DD;

    __shared__ __bf16 sPh[128 * LDS_STRIDE];
    __shared__ __bf16 sPl[128 * LDS_STRIDE];
    __shared__ __bf16 sVh[64 * LDS_STRIDE];   // V^T: row = d, col = j (32)
    __shared__ __bf16 sVl[64 * LDS_STRIDE];

    const int tid = threadIdx.x, lane = tid & 31, wave = tid >> 5;

    v8f acc[4];
#pragma unroll
    for (int j = 0; j < 4; j++)
#pragma unroll
        for (int r = 0; r < 8; r++) acc[j][r] = 0.0f;

    // V chunk prefetch: 32x64 floats, 8 scalars/thread (coalesced in d)
    const int vj = tid >> 5;          // reuse wave id: rows vj, vj+8, ... vj+24? no:
    // thread handles idx = tid + t*256 -> jj = idx>>6, d = idx&63
    float4 pp[4];
    float  pv[8];

    auto v_load = [&](int j0) {
#pragma unroll
        for (int t = 0; t < 8; t++) {
            int idx = tid + (t << 8);
            int jj = idx >> 6, d = idx & 63;
            pv[t] = V[(size_t)(j0 + jj) * DD + d];
        }
    };
    auto v_store = [&]() {
#pragma unroll
        for (int t = 0; t < 8; t++) {
            int idx = tid + (t << 8);
            int jj = idx >> 6, d = idx & 63;
            __bf16 hh, ll;
            cvt_hilo(pv[t], hh, ll);
            int o = d * LDS_STRIDE + jj;
            sVh[o] = hh;
            sVl[o] = ll;
        }
    };

    const int jmax = (it + 1) * 128;
    tile_load_g(P, LL, tid, pp);
    v_load(0);
    tile_cvt_store(pp, sPh, sPl, tid);
    v_store();
    __syncthreads();

    for (int j0 = 0; j0 < jmax; j0 += 32) {
        const bool hasNext = (j0 + 32) < jmax;
        if (hasNext) {
            tile_load_g(P + j0 + 32, LL, tid, pp);
            v_load(j0 + 32);
        }
        v16bf pah = frag_ld_a(sPh, wave * 16, lane);
        v16bf pal = frag_ld_a(sPl, wave * 16, lane);
#pragma unroll
        for (int nj = 0; nj < 4; nj++) {
            v16bf vbh = frag_ld_b(sVh, nj * 16, lane);
            v16bf vbl = frag_ld_b(sVl, nj * 16, lane);
            acc[nj] = wmma_bf16(pah, vbh, acc[nj]);
            acc[nj] = wmma_bf16(pah, vbl, acc[nj]);
            acc[nj] = wmma_bf16(pal, vbh, acc[nj]);
        }
        __syncthreads();
        if (hasNext) {
            tile_cvt_store(pp, sPh, sPl, tid);
            v_store();
        }
        __syncthreads();
    }

    const int lcol = lane & 15;
    const int rbase = (lane & 16) ? 8 : 0;
#pragma unroll
    for (int nj = 0; nj < 4; nj++)
#pragma unroll
        for (int r = 0; r < 8; r++) {
            int lq = it * 128 + wave * 16 + rbase + r;
            int d = nj * 16 + lcol;
            OH[((size_t)b * LL + lq) * EE + h * DD + d] = acc[nj][r];
        }
}

// ---------------------------------------------------------------------------
extern "C" void kernel_launch(void* const* d_in, const int* in_sizes, int n_in,
                              void* d_out, int out_size, void* d_ws, size_t ws_size,
                              hipStream_t stream) {
    const float* q  = (const float*)d_in[0];
    const float* k  = (const float*)d_in[1];
    const float* v  = (const float*)d_in[2];
    // d_in[3] = causal mask (implicit in kernels; unused)
    const float* Wq = (const float*)d_in[4];
    const float* Wk = (const float*)d_in[5];
    const float* Wv = (const float*)d_in[6];
    const float* Wo = (const float*)d_in[7];

    float* out  = (float*)d_out;
    float* attn = out + (size_t)BB * LL * EE;  // second tuple output

    const size_t PROJ = (size_t)BB * HH * LL * DD;  // 8.39M floats
    float* Qb = (float*)d_ws;
    float* Kb = Qb + PROJ;
    float* Vb = Kb + PROJ;
    float* OH = Qb;  // safe reuse: Q dead after scores kernel

    const int M = BB * LL;  // 8192
    dim3 gproj(M / 128, EE / 128);  // 64 x 8

    gemm_xwT_kernel<<<gproj, 256, 0, stream>>>(q, Wq, Qb, M, EE, 1);
    gemm_xwT_kernel<<<gproj, 256, 0, stream>>>(k, Wk, Kb, M, EE, 1);
    gemm_xwT_kernel<<<gproj, 256, 0, stream>>>(v, Wv, Vb, M, EE, 1);

    attn_scores_kernel<<<dim3(LL / 128, LL / 128, BB * HH), 256, 0, stream>>>(Qb, Kb, attn);
    softmax_kernel<<<dim3(BB * HH * LL), 256, 0, stream>>>(attn);
    attn_av_kernel<<<dim3(LL / 128, BB * HH), 256, 0, stream>>>(attn, Vb, OH);

    gemm_xwT_kernel<<<gproj, 256, 0, stream>>>(OH, Wo, out, M, EE, 0);
}
